// MoeMLPForLoop_16655883174265
// MI455X (gfx1250) — matrix operand order repaired and checked by reading, not compiled, not run on hardware
//
#include <hip/hip_runtime.h>
#include <hip/hip_bf16.h>
#include <math.h>

#define N_TOK 8192
#define DIM   1024
#define NEXP  8
#define DFFN  4096

#define LDST  40     // padded LDS stride in ushorts (80B rows -> conflict-free frag loads)
#define TM    256
#define TN    128

typedef __attribute__((ext_vector_type(16))) __bf16 v16bf;
typedef __attribute__((ext_vector_type(8)))  float  v8f;

union Frag16 { v16bf v; unsigned int u[8]; };
union Acc    { v8f   v; float        f[8]; };

// ---- workspace layout (bytes) ----
#define OFF_COUNTS 0
#define OFF_LIST   1024
#define OFF_WT     (OFF_LIST + NEXP * N_TOK * 4)
#define OFF_XBF    (OFF_WT   + NEXP * N_TOK * 4)
#define OFF_H      (OFF_XBF  + (size_t)N_TOK * DIM * 2)
#define OFF_WFCT   (OFF_H    + (size_t)N_TOK * DFFN * 2)                 // bf16 w_fc^T  [E][DFFN][DIM]
#define OFF_WPJT   (OFF_WFCT + (size_t)NEXP * DIM * DFFN * 2)            // bf16 w_proj^T [E][DIM][DFFN]
#define WS_PRE_REQ (OFF_WPJT + (size_t)NEXP * DIM * DFFN * 2)            // ~219 MB
#define WS_MIN_REQ OFF_WFCT                                              // ~84 MB

__device__ __forceinline__ unsigned short f2bf(float f) {
  unsigned int u = __float_as_uint(f);
  unsigned int r = u + 0x7FFFu + ((u >> 16) & 1u);   // round-to-nearest-even
  return (unsigned short)(r >> 16);
}

__device__ __forceinline__ float gelu_exact(float v) {
  return 0.5f * v * (1.0f + erff(v * 0.70710678118654752f));
}

// low 32 bits of the flat shared-aperture address == LDS offset (ISA 10.2)
__device__ __forceinline__ unsigned ldsoff(const void* p) {
  return (unsigned)(unsigned long long)p;
}
// CDNA5 async global->LDS copy, tracked by ASYNCcnt
__device__ __forceinline__ void async_b128(unsigned lds, const void* g) {
  asm volatile("global_load_async_to_lds_b128 %0, %1, off"
               :: "v"(lds), "v"((unsigned long long)g) : "memory");
}
__device__ __forceinline__ void wait_async() {
  asm volatile("s_wait_asynccnt 0x0" ::: "memory");
}

// A fragment: 16x32 bf16; vgpr j<4: K=2j+{0,1}+8*hi ; j>=4: K=16+2(j-4)+{0,1}+8*hi
__device__ __forceinline__ void load_frag_a(const unsigned short* strip, int lane, Frag16& f) {
  int m = lane & 15, hi = lane >> 4;
  const unsigned short* base = strip + m * LDST;
#pragma unroll
  for (int j = 0; j < 8; ++j) {
    int k = (j < 4) ? (2 * j + 8 * hi) : (2 * (j - 4) + 16 + 8 * hi);
    f.u[j] = *(const unsigned int*)(base + k);
  }
}
// B fragment: 32x16 bf16; vgpr j<4: K=2j+{0,1}+8*hi ; j>=4: K=16+2(j-4)+{0,1}+8*hi
__device__ __forceinline__ void load_frag_b(const unsigned short* strip, int lane, Frag16& f) {
  int n = lane & 15, hi = lane >> 4;
  const unsigned short* base = strip + n * LDST;
#pragma unroll
  for (int j = 0; j < 8; ++j) {
    int k = (j < 4) ? (2 * j + 8 * hi) : (2 * (j - 4) + 16 + 8 * hi);
    f.u[j] = *(const unsigned int*)(base + k);
  }
}

__global__ void init_kernel(float* out, int* counts) {
  long long idx = (long long)blockIdx.x * blockDim.x + threadIdx.x;
  if (idx < (long long)N_TOK * DIM / 4)
    ((float4*)out)[idx] = make_float4(0.f, 0.f, 0.f, 0.f);
  if (blockIdx.x == 0 && threadIdx.x < NEXP) counts[threadIdx.x] = 0;
}

__global__ void cvt_x_kernel(const float* __restrict__ x, unsigned short* __restrict__ xbf) {
  long long i = ((long long)blockIdx.x * blockDim.x + threadIdx.x) * 4;
  float4 v = *(const float4*)(x + i);
  union { unsigned short s[4]; uint2 u; } p;
  p.s[0] = f2bf(v.x); p.s[1] = f2bf(v.y); p.s[2] = f2bf(v.z); p.s[3] = f2bf(v.w);
  *(uint2*)(xbf + i) = p.u;
}

// fp32 [R][C] -> bf16 transposed [C][R], 32x32 LDS tile transpose
__global__ __launch_bounds__(256)
void cvtT_kernel(const float* __restrict__ src, unsigned short* __restrict__ dst, int R, int C) {
  __shared__ unsigned short t[32][36];
  int r0 = blockIdx.y * 32, c0 = blockIdx.x * 32;
  int tid = threadIdx.x;
  int rl = tid >> 3, c4 = (tid & 7) * 4;
  float4 v = *(const float4*)(src + (long long)(r0 + rl) * C + c0 + c4);
  t[rl][c4 + 0] = f2bf(v.x); t[rl][c4 + 1] = f2bf(v.y);
  t[rl][c4 + 2] = f2bf(v.z); t[rl][c4 + 3] = f2bf(v.w);
  __syncthreads();
  union { unsigned short s[4]; uint2 u; } o;
#pragma unroll
  for (int j = 0; j < 4; ++j) o.s[j] = t[c4 + j][rl];
  *(uint2*)(dst + (long long)(c0 + rl) * R + r0 + c4) = o.u;
}

// one wave32 per token
__global__ void router_kernel(const float* __restrict__ x, const float* __restrict__ rw,
                              float* __restrict__ logits, int* counts,
                              int* __restrict__ list, float* __restrict__ wts) {
  int wave = (int)((blockIdx.x * blockDim.x + threadIdx.x) >> 5);
  int lane = threadIdx.x & 31;
  if (wave >= N_TOK) return;
  const float* xr = x + (long long)wave * DIM;
  float acc[NEXP];
#pragma unroll
  for (int e = 0; e < NEXP; ++e) acc[e] = 0.f;
  for (int i = 0; i < DIM / 32; ++i) {
    int d = i * 32 + lane;
    float xv = xr[d];
    const float* r = rw + d * NEXP;
#pragma unroll
    for (int e = 0; e < NEXP; ++e) acc[e] += xv * r[e];
  }
#pragma unroll
  for (int off = 16; off > 0; off >>= 1)
#pragma unroll
    for (int e = 0; e < NEXP; ++e) acc[e] += __shfl_xor(acc[e], off, 32);
  if (lane == 0) {
    float m = acc[0];
#pragma unroll
    for (int e = 1; e < NEXP; ++e) m = fmaxf(m, acc[e]);
    float p[NEXP];
#pragma unroll
    for (int e = 0; e < NEXP; ++e) { logits[(long long)wave * NEXP + e] = acc[e]; p[e] = __expf(acc[e] - m); }
    int i0 = 0;
#pragma unroll
    for (int e = 1; e < NEXP; ++e) if (p[e] > p[i0]) i0 = e;
    int i1 = (i0 == 0) ? 1 : 0;
#pragma unroll
    for (int e = 0; e < NEXP; ++e) if (e != i0 && p[e] > p[i1]) i1 = e;
    float s = p[i0] + p[i1];
    float w0 = p[i0] / s, w1 = p[i1] / s;
    int p0 = atomicAdd(&counts[i0], 1);
    list[i0 * N_TOK + p0] = wave; wts[i0 * N_TOK + p0] = w0;
    int p1 = atomicAdd(&counts[i1], 1);
    list[i1 * N_TOK + p1] = wave; wts[i1 * N_TOK + p1] = w1;
  }
}

// H[pos,:] = gelu( x[list[pos],:] @ w_fc[e] );  256x128 tiles, K-step 32, double-buffered
template <bool PRE>
__global__ __launch_bounds__(256)
void gemm1_kernel(const unsigned short* __restrict__ xbf,
                  const unsigned short* __restrict__ wT,   // PRE: bf16 w_fc^T [E][DFFN][DIM]
                  const float* __restrict__ wf,            // !PRE: fp32 w_fc  [E][DIM][DFFN]
                  const int* __restrict__ counts, const int* __restrict__ list,
                  unsigned short* __restrict__ H, int e) {
  __shared__ unsigned short As[2][TM * LDST];
  __shared__ unsigned short Bs[2][TN * LDST];
  int cnt = counts[e];
  int row0 = blockIdx.y * TM;
  if (row0 >= cnt) return;
  int col0 = blockIdx.x * TN;
  int tid = threadIdx.x, lane = tid & 31, wid = tid >> 5;

  const int* lst = list + e * N_TOK;
  int rpos = row0 + tid; if (rpos > cnt - 1) rpos = cnt - 1;
  const unsigned short* arow_g = xbf + (long long)lst[rpos] * DIM;

  const unsigned short* bT_g =
      PRE ? wT + (long long)e * DIM * DFFN + (long long)(col0 + (tid >> 1)) * DIM + (tid & 1) * 16
          : nullptr;
  const float* bF_g = PRE ? nullptr : wf + (long long)e * DIM * DFFN + col0;

  Acc c0[8], c1[8];
#pragma unroll
  for (int t = 0; t < 8; ++t)
#pragma unroll
    for (int r = 0; r < 8; ++r) { c0[t].f[r] = 0.f; c1[t].f[r] = 0.f; }

  auto stage = [&](int buf, int k0) {
#pragma unroll
    for (int p = 0; p < 2; ++p)
      async_b128(ldsoff(&As[buf][tid * LDST + p * 16]), (const void*)(arow_g + k0 + p * 16));
    if (PRE) {
      async_b128(ldsoff(&Bs[buf][(tid >> 1) * LDST + (tid & 1) * 16]), (const void*)(bT_g + k0));
    } else {
#pragma unroll
      for (int i = 0; i < 4; ++i) {
        int q = tid + i * 256;
        int dl = q >> 5, f4 = q & 31;
        float4 wv = *(const float4*)(bF_g + (long long)(k0 + dl) * DFFN + f4 * 4);
        int fl = f4 * 4;
        Bs[buf][(fl + 0) * LDST + dl] = f2bf(wv.x);
        Bs[buf][(fl + 1) * LDST + dl] = f2bf(wv.y);
        Bs[buf][(fl + 2) * LDST + dl] = f2bf(wv.z);
        Bs[buf][(fl + 3) * LDST + dl] = f2bf(wv.w);
      }
    }
  };
  auto compute = [&](int buf) {
    Frag16 a0, a1;
    load_frag_a(&As[buf][(wid * 32) * LDST], lane, a0);
    load_frag_a(&As[buf][(wid * 32 + 16) * LDST], lane, a1);
#pragma unroll
    for (int ct = 0; ct < 8; ++ct) {
      Frag16 b; load_frag_b(&Bs[buf][ct * 16 * LDST], lane, b);
      c0[ct].v = __builtin_amdgcn_wmma_f32_16x16x32_bf16(false, a0.v, false, b.v,
                                                         (short)0, c0[ct].v, false, false);
      c1[ct].v = __builtin_amdgcn_wmma_f32_16x16x32_bf16(false, a1.v, false, b.v,
                                                         (short)0, c1[ct].v, false, false);
    }
  };

  stage(0, 0);
  const int NK = DIM / 32;
  for (int i = 0; i < NK; ++i) {
    wait_async();
    __syncthreads();
    if (i + 1 < NK) stage((i + 1) & 1, (i + 1) * 32);
    compute(i & 1);
  }

  int n = lane & 15, hi = lane >> 4;
#pragma unroll
  for (int s = 0; s < 2; ++s) {
    Acc* cc = s ? c1 : c0;
#pragma unroll
    for (int ct = 0; ct < 8; ++ct)
#pragma unroll
      for (int r = 0; r < 8; ++r) {
        int row = row0 + wid * 32 + s * 16 + r + 8 * hi;
        if (row < cnt)
          H[(long long)row * DFFN + col0 + ct * 16 + n] = f2bf(gelu_exact(cc[ct].f[r]));
      }
  }
}

// out[list[pos],:] += wts[pos] * ( H[pos,:] @ w_proj[e] )
template <bool PRE>
__global__ __launch_bounds__(256)
void gemm2_kernel(const unsigned short* __restrict__ H,
                  const unsigned short* __restrict__ wT,   // PRE: bf16 w_proj^T [E][DIM][DFFN]
                  const float* __restrict__ wf,            // !PRE: fp32 w_proj  [E][DFFN][DIM]
                  const int* __restrict__ counts, const int* __restrict__ list,
                  const float* __restrict__ wts, float* __restrict__ out, int e) {
  __shared__ unsigned short As[2][TM * LDST];
  __shared__ unsigned short Bs[2][TN * LDST];
  int cnt = counts[e];
  int row0 = blockIdx.y * TM;
  if (row0 >= cnt) return;
  int col0 = blockIdx.x * TN;
  int tid = threadIdx.x, lane = tid & 31, wid = tid >> 5;

  const unsigned short* arow_g = H + (long long)(row0 + tid) * DFFN;
  const unsigned short* bT_g =
      PRE ? wT + (long long)e * DIM * DFFN + (long long)(col0 + (tid >> 1)) * DFFN + (tid & 1) * 16
          : nullptr;
  const float* bF_g = PRE ? nullptr : wf + (long long)e * DFFN * DIM + col0;

  Acc c0[8], c1[8];
#pragma unroll
  for (int t = 0; t < 8; ++t)
#pragma unroll
    for (int r = 0; r < 8; ++r) { c0[t].f[r] = 0.f; c1[t].f[r] = 0.f; }

  auto stage = [&](int buf, int k0) {
#pragma unroll
    for (int p = 0; p < 2; ++p)
      async_b128(ldsoff(&As[buf][tid * LDST + p * 16]), (const void*)(arow_g + k0 + p * 16));
    if (PRE) {
      async_b128(ldsoff(&Bs[buf][(tid >> 1) * LDST + (tid & 1) * 16]), (const void*)(bT_g + k0));
    } else {
#pragma unroll
      for (int i = 0; i < 4; ++i) {
        int q = tid + i * 256;
        int fl = q >> 5, c4 = q & 31;
        float4 wv = *(const float4*)(bF_g + (long long)(k0 + fl) * DIM + c4 * 4);
        int cl = c4 * 4;
        Bs[buf][(cl + 0) * LDST + fl] = f2bf(wv.x);
        Bs[buf][(cl + 1) * LDST + fl] = f2bf(wv.y);
        Bs[buf][(cl + 2) * LDST + fl] = f2bf(wv.z);
        Bs[buf][(cl + 3) * LDST + fl] = f2bf(wv.w);
      }
    }
  };
  auto compute = [&](int buf) {
    Frag16 a0, a1;
    load_frag_a(&As[buf][(wid * 32) * LDST], lane, a0);
    load_frag_a(&As[buf][(wid * 32 + 16) * LDST], lane, a1);
#pragma unroll
    for (int ct = 0; ct < 8; ++ct) {
      Frag16 b; load_frag_b(&Bs[buf][ct * 16 * LDST], lane, b);
      c0[ct].v = __builtin_amdgcn_wmma_f32_16x16x32_bf16(false, a0.v, false, b.v,
                                                         (short)0, c0[ct].v, false, false);
      c1[ct].v = __builtin_amdgcn_wmma_f32_16x16x32_bf16(false, a1.v, false, b.v,
                                                         (short)0, c1[ct].v, false, false);
    }
  };

  stage(0, 0);
  const int NK = DFFN / 32;
  for (int i = 0; i < NK; ++i) {
    wait_async();
    __syncthreads();
    if (i + 1 < NK) stage((i + 1) & 1, (i + 1) * 32);
    compute(i & 1);
  }

  int n = lane & 15, hi = lane >> 4;
#pragma unroll
  for (int s = 0; s < 2; ++s) {
    Acc* cc = s ? c1 : c0;
#pragma unroll
    for (int r = 0; r < 8; ++r) {
      int row = row0 + wid * 32 + s * 16 + r + 8 * hi;
      if (row < cnt) {
        int tok = list[e * N_TOK + row];
        float w = wts[e * N_TOK + row];
        float* orow = out + (long long)tok * DIM + col0 + n;
#pragma unroll
        for (int ct = 0; ct < 8; ++ct)
          orow[ct * 16] += w * cc[ct].f[r];   // one writer per element per launch
      }
    }
  }
}

extern "C" void kernel_launch(void* const* d_in, const int* in_sizes, int n_in,
                              void* d_out, int out_size, void* d_ws, size_t ws_size,
                              hipStream_t stream) {
  const float* x   = (const float*)d_in[0];
  const float* rw  = (const float*)d_in[1];
  const float* wfc = (const float*)d_in[2];
  const float* wpj = (const float*)d_in[3];
  float* out    = (float*)d_out;
  float* logits = out + (size_t)N_TOK * DIM;

  char* ws = (char*)d_ws;
  int*            counts = (int*)(ws + OFF_COUNTS);
  int*            list   = (int*)(ws + OFF_LIST);
  float*          wts    = (float*)(ws + OFF_WT);
  unsigned short* xbf    = (unsigned short*)(ws + OFF_XBF);
  unsigned short* Hbuf   = (unsigned short*)(ws + OFF_H);
  unsigned short* wfcT   = (unsigned short*)(ws + OFF_WFCT);
  unsigned short* wpjT   = (unsigned short*)(ws + OFF_WPJT);

  const bool pre = (ws_size >= WS_PRE_REQ);   // deterministic: ws_size fixed per run

  init_kernel<<<(N_TOK * DIM / 4 + 255) / 256, 256, 0, stream>>>(out, counts);
  cvt_x_kernel<<<N_TOK * DIM / 4 / 256, 256, 0, stream>>>(x, xbf);
  router_kernel<<<N_TOK / 8, 256, 0, stream>>>(x, rw, logits, counts, list, wts);

  if (pre) {
    for (int e = 0; e < NEXP; ++e) {
      cvtT_kernel<<<dim3(DFFN / 32, DIM / 32), 256, 0, stream>>>(
          wfc + (size_t)e * DIM * DFFN, wfcT + (size_t)e * DIM * DFFN, DIM, DFFN);
      cvtT_kernel<<<dim3(DIM / 32, DFFN / 32), 256, 0, stream>>>(
          wpj + (size_t)e * DFFN * DIM, wpjT + (size_t)e * DIM * DFFN, DFFN, DIM);
    }
  }

  for (int e = 0; e < NEXP; ++e) {
    if (pre) {
      gemm1_kernel<true><<<dim3(DFFN / TN, N_TOK / TM), 256, 0, stream>>>(
          xbf, wfcT, nullptr, counts, list, Hbuf, e);
      gemm2_kernel<true><<<dim3(DIM / TN, N_TOK / TM), 256, 0, stream>>>(
          Hbuf, wpjT, nullptr, counts, list, wts, out, e);
    } else {
      gemm1_kernel<false><<<dim3(DFFN / TN, N_TOK / TM), 256, 0, stream>>>(
          xbf, nullptr, wfc, counts, list, Hbuf, e);
      gemm2_kernel<false><<<dim3(DIM / TN, N_TOK / TM), 256, 0, stream>>>(
          Hbuf, nullptr, wpj, counts, list, wts, out, e);
    }
  }
}
